// GNNInitializer_51539608059
// MI455X (gfx1250) — matrix operand order repaired and checked by reading, not compiled, not run on hardware
//
#include <hip/hip_runtime.h>

typedef __attribute__((ext_vector_type(16))) __bf16 v16bf;
typedef __attribute__((ext_vector_type(8)))  __bf16 v8bf;
typedef __attribute__((ext_vector_type(8)))  float  v8f;

#define NNODES 50000
#define DD 16          // neighbors / LSTM steps
#define FF 128         // feature dim
#define GF 512         // 4*F gate dim
#define NTILES (NNODES / 16)

__device__ __forceinline__ unsigned short f2bf(float f) {
    union { float f; unsigned u; } x; x.f = f;
    unsigned u = x.u;
    return (unsigned short)((u + 0x7FFFu + ((u >> 16) & 1u)) >> 16);  // RNE
}
__device__ __forceinline__ float sigm(float x) { return 1.0f / (1.0f + __expf(-x)); }
__device__ __forceinline__ float tanh_fast(float x) {
    float e = __expf(2.0f * x);
    return (e - 1.0f) / (e + 1.0f);
}

// Convert weights to bf16 once. Wih/Whh kept row-major [4F][F] (B = W^T is then
// per-lane contiguous along K). Wself/Wneigh stored TRANSPOSED [j][k] so the
// non-transposed GEMM x@W also gets contiguous B-fragment loads.
__global__ void convert_weights_kernel(const float* __restrict__ Wih,
                                       const float* __restrict__ Whh,
                                       const float* __restrict__ Wself,
                                       const float* __restrict__ Wneigh,
                                       unsigned short* __restrict__ wih_bf,
                                       unsigned short* __restrict__ whh_bf,
                                       unsigned short* __restrict__ wsfT_bf,
                                       unsigned short* __restrict__ wngT_bf) {
    int tid = blockIdx.x * blockDim.x + threadIdx.x;
    if (tid < 3 * GF * FF) {
        wih_bf[tid] = f2bf(Wih[tid]);
        whh_bf[tid] = f2bf(Whh[tid]);
    }
    if (tid < 3 * FF * FF) {
        int l = tid / (FF * FF);
        int r = tid % (FF * FF);
        int j = r / FF, k = r % FF;
        wsfT_bf[tid] = f2bf(Wself [l * FF * FF + k * FF + j]);
        wngT_bf[tid] = f2bf(Wneigh[l * FF * FF + k * FF + j]);
    }
}

union Frag { v16bf v; v8bf h[2]; };

__global__ __launch_bounds__(512, 1)
void sage_lstm_layer_kernel(const float* __restrict__ xin,
                            const int*   __restrict__ nidx,
                            const unsigned short* __restrict__ wih,    // [512][128] bf16
                            const unsigned short* __restrict__ whh,    // [512][128] bf16
                            const unsigned short* __restrict__ wsfT,   // [128][128] bf16 (transposed)
                            const unsigned short* __restrict__ wngT,   // [128][128] bf16 (transposed)
                            const float* __restrict__ bih,             // [512]
                            const float* __restrict__ bhh,             // [512]
                            const float* __restrict__ bout,            // [128]
                            float* __restrict__ out) {
    __shared__ alignas(16) unsigned short m_sh[16 * FF];  // step input (bf16)
    __shared__ alignas(16) unsigned short h_sh[16 * FF];  // hidden state (bf16)
    __shared__ alignas(16) float g_sh[16 * GF];           // gate accumulators (f32)
    __shared__ alignas(16) float c_sh[16 * FF];           // cell state (f32)
    __shared__ alignas(16) int   idx_sh[16 * DD];         // this tile's neighbor indices

    const int tid  = threadIdx.x;
    const int lane = tid & 31;
    const int wv   = tid >> 5;       // 0..15 waves
    const int l16  = lane & 15;
    const int hi   = lane >> 4;      // 0 or 1 (half-wave)
    const int tile = blockIdx.x;

    // zero h and c; stage the tile's 16x16 neighbor-index block into LDS
    for (int e = tid; e < 16 * FF; e += 512) {
        h_sh[e] = 0;
        c_sh[e] = 0.0f;
    }
    if (tid < 16 * DD) idx_sh[tid] = nidx[tile * (16 * DD) + tid];

    // ---- Preload this wave's B-fragments for Wih/Whh into registers ----
    // B (K x 16, bf16): lane = N%16 column; lanes 0-15 hold K 0-7,16-23, lanes 16-31
    // hold K 8-15,24-31 of each 32-deep K chunk (matches A layout).
    const int kb = hi ? 8 : 0;
    v16bf Bih[2][4], Bhh[2][4];
#pragma unroll
    for (int nt = 0; nt < 2; ++nt) {
        int row = (wv * 2 + nt) * 16 + l16;           // gate output index 0..511
        const unsigned short* pih = wih + row * FF;
        const unsigned short* phh = whh + row * FF;
#pragma unroll
        for (int kc = 0; kc < 4; ++kc) {
            int k0 = kc * 32 + kb;
            Frag b;
            b.h[0] = *reinterpret_cast<const v8bf*>(pih + k0);
            b.h[1] = *reinterpret_cast<const v8bf*>(pih + k0 + 16);
            Bih[nt][kc] = b.v;
            b.h[0] = *reinterpret_cast<const v8bf*>(phh + k0);
            b.h[1] = *reinterpret_cast<const v8bf*>(phh + k0 + 16);
            Bhh[nt][kc] = b.v;
        }
    }

    // Per-thread fixed bias sums for the elementwise cell update (hoisted).
    float bsi[4], bsf[4], bsg[4], bso[4];
#pragma unroll
    for (int q = 0; q < 4; ++q) {
        int f = (tid * 4 + q) & 127;
        bsi[q] = bih[f]       + bhh[f];
        bsf[q] = bih[128 + f] + bhh[128 + f];
        bsg[q] = bih[256 + f] + bhh[256 + f];
        bso[q] = bih[384 + f] + bhh[384 + f];
    }

    // A (16 x 32 bf16): lane M = lane%16; lanes 0-15: elems 0-7 = K 0-7,
    // elems 8-15 = K 16-23; lanes 16-31: K 8-15 and K 24-31.
    const int am = l16;
    const int ak = hi ? 8 : 0;

    // Gather coordinates for staging (fixed per thread).
    const int sr   = tid >> 5;            // node row 0..15
    const int sc4  = (tid & 31) << 2;     // feature col, step 4
    const int node = tile * 16 + sr;

    __syncthreads();

    // Prefetch step 0's gathered neighbor row (index from LDS, no global dep chain).
    float4 pref = *reinterpret_cast<const float4*>(
        xin + (size_t)idx_sh[sr * DD + 0] * FF + sc4);

    // ---------------- LSTM over the 16 gathered neighbors ----------------
    for (int t = 0; t < DD; ++t) {
        // commit prefetched neighbor row for step t (f32 -> bf16 into LDS)
        m_sh[sr * FF + sc4 + 0] = f2bf(pref.x);
        m_sh[sr * FF + sc4 + 1] = f2bf(pref.y);
        m_sh[sr * FF + sc4 + 2] = f2bf(pref.z);
        m_sh[sr * FF + sc4 + 3] = f2bf(pref.w);
        // issue prefetch for step t+1 (consumed after three more barriers)
        if (t + 1 < DD) {
            pref = *reinterpret_cast<const float4*>(
                xin + (size_t)idx_sh[sr * DD + t + 1] * FF + sc4);
        }
        __syncthreads();

        // gates[16, 512] = m_t @ Wih^T + h @ Whh^T  (this wave: 2 x 16 columns)
        // Load ALL A fragments, fence the scheduler, then 16 WMMAs back-to-back.
        Frag am_[4], ah_[4];
#pragma unroll
        for (int kc = 0; kc < 4; ++kc) {
            int base = am * FF + kc * 32 + ak;
            am_[kc].h[0] = *reinterpret_cast<const v8bf*>(m_sh + base);
            am_[kc].h[1] = *reinterpret_cast<const v8bf*>(m_sh + base + 16);
            ah_[kc].h[0] = *reinterpret_cast<const v8bf*>(h_sh + base);
            ah_[kc].h[1] = *reinterpret_cast<const v8bf*>(h_sh + base + 16);
        }
        __builtin_amdgcn_sched_barrier(0);   // keep ds_loads batched ahead of WMMAs
        v8f acc0 = {}, acc1 = {};
#pragma unroll
        for (int kc = 0; kc < 4; ++kc) {
            acc0 = __builtin_amdgcn_wmma_f32_16x16x32_bf16(
                     false, am_[kc].v, false, Bih[0][kc], (short)0, acc0, false, false);
            acc1 = __builtin_amdgcn_wmma_f32_16x16x32_bf16(
                     false, am_[kc].v, false, Bih[1][kc], (short)0, acc1, false, false);
        }
#pragma unroll
        for (int kc = 0; kc < 4; ++kc) {
            acc0 = __builtin_amdgcn_wmma_f32_16x16x32_bf16(
                     false, ah_[kc].v, false, Bhh[0][kc], (short)0, acc0, false, false);
            acc1 = __builtin_amdgcn_wmma_f32_16x16x32_bf16(
                     false, ah_[kc].v, false, Bhh[1][kc], (short)0, acc1, false, false);
        }
        // D layout: VGPR v -> M = v + 8*hi, N = lane%16
        {
            int cb0 = (wv * 2 + 0) * 16 + l16;
            int cb1 = (wv * 2 + 1) * 16 + l16;
#pragma unroll
            for (int v = 0; v < 8; ++v) {
                g_sh[(v + 8 * hi) * GF + cb0] = acc0[v];
                g_sh[(v + 8 * hi) * GF + cb1] = acc1[v];
            }
        }
        __syncthreads();

        // elementwise LSTM cell update: 512 threads x 4 elems over [16,128]
#pragma unroll
        for (int q = 0; q < 4; ++q) {
            int e = tid * 4 + q;
            int M = e >> 7, f = e & 127;
            const float* gr = g_sh + M * GF;
            float gi = gr[f]       + bsi[q];
            float gf = gr[128 + f] + bsf[q];
            float gg = gr[256 + f] + bsg[q];
            float go = gr[384 + f] + bso[q];
            float cz = sigm(gf) * c_sh[M * FF + f] + sigm(gi) * tanh_fast(gg);
            float hz = sigm(go) * tanh_fast(cz);
            c_sh[M * FF + f] = cz;
            h_sh[M * FF + f] = f2bf(hz);
        }
        __syncthreads();
    }

    // ---------------- SAGE combine: relu(x@Wself + h@Wneigh + b) ----------------
    // stage this tile's own node features into m_sh (bf16)
    {
        const float4 v = *reinterpret_cast<const float4*>(
            xin + (size_t)node * FF + sc4);
        m_sh[sr * FF + sc4 + 0] = f2bf(v.x);
        m_sh[sr * FF + sc4 + 1] = f2bf(v.y);
        m_sh[sr * FF + sc4 + 2] = f2bf(v.z);
        m_sh[sr * FF + sc4 + 3] = f2bf(v.w);
    }
    __syncthreads();

    {
        int sel = wv >> 3;                 // 0: x@Wself, 1: h@Wneigh
        int jt  = wv & 7;                  // output N-tile 0..7
        const unsigned short* Bp = (sel ? wngT : wsfT) + (jt * 16 + l16) * FF;
        const unsigned short* Ap = sel ? h_sh : m_sh;
        Frag af[4], bf[4];
#pragma unroll
        for (int kc = 0; kc < 4; ++kc) {
            int base = am * FF + kc * 32 + ak;
            af[kc].h[0] = *reinterpret_cast<const v8bf*>(Ap + base);
            af[kc].h[1] = *reinterpret_cast<const v8bf*>(Ap + base + 16);
            int k0 = kc * 32 + kb;
            bf[kc].h[0] = *reinterpret_cast<const v8bf*>(Bp + k0);
            bf[kc].h[1] = *reinterpret_cast<const v8bf*>(Bp + k0 + 16);
        }
        __builtin_amdgcn_sched_barrier(0);
        v8f acc = {};
#pragma unroll
        for (int kc = 0; kc < 4; ++kc) {
            acc = __builtin_amdgcn_wmma_f32_16x16x32_bf16(
                    false, af[kc].v, false, bf[kc].v, (short)0, acc, false, false);
        }
        int cb = sel * 128 + jt * 16 + l16;   // two partial results in g_sh cols [0,256)
#pragma unroll
        for (int v = 0; v < 8; ++v)
            g_sh[(v + 8 * hi) * GF + cb] = acc[v];
    }
    __syncthreads();

    // sum the two partials, add bias, relu, store fp32
    {
        float4 o;
        float* op = &o.x;
#pragma unroll
        for (int j = 0; j < 4; ++j) {
            float v = g_sh[sr * GF + sc4 + j] + g_sh[sr * GF + 128 + sc4 + j] + bout[sc4 + j];
            op[j] = v > 0.0f ? v : 0.0f;
        }
        *reinterpret_cast<float4*>(out + (size_t)node * FF + sc4) = o;
    }
}

extern "C" void kernel_launch(void* const* d_in, const int* in_sizes, int n_in,
                              void* d_out, int out_size, void* d_ws, size_t ws_size,
                              hipStream_t stream) {
    const float* x      = (const float*)d_in[0];
    const int*   nidx   = (const int*)  d_in[1];
    const float* Wih    = (const float*)d_in[2];
    const float* Whh    = (const float*)d_in[3];
    const float* bih    = (const float*)d_in[4];
    const float* bhh    = (const float*)d_in[5];
    const float* Wself  = (const float*)d_in[6];
    const float* Wneigh = (const float*)d_in[7];
    const float* bb     = (const float*)d_in[8];
    float* out = (float*)d_out;

    char* ws = (char*)d_ws;
    unsigned short* wih_bf = (unsigned short*)(ws);                    // 3*512*128*2 = 393216 B
    unsigned short* whh_bf = (unsigned short*)(ws + 393216);           // 393216 B
    unsigned short* wsf_bf = (unsigned short*)(ws + 786432);           // 3*128*128*2 = 98304 B
    unsigned short* wng_bf = (unsigned short*)(ws + 884736);           // 98304 B
    float*          h_ws   = (float*)(ws + 1048576);                   // 50000*128*4 = 25.6 MB

    convert_weights_kernel<<<(3 * GF * FF + 255) / 256, 256, 0, stream>>>(
        Wih, Whh, Wself, Wneigh, wih_bf, whh_bf, wsf_bf, wng_bf);

    const float* lin[3] = { x, out, h_ws };
    float*      lout[3] = { out, h_ws, out };
    for (int l = 0; l < 3; ++l) {
        sage_lstm_layer_kernel<<<NTILES, 512, 0, stream>>>(
            lin[l], nidx,
            wih_bf + l * GF * FF, whh_bf + l * GF * FF,
            wsf_bf + l * FF * FF, wng_bf + l * FF * FF,
            bih + l * GF, bhh + l * GF, bb + l * FF,
            lout[l]);
    }
}
